// SDFVae_20478404067352
// MI455X (gfx1250) — matrix operand order repaired and checked by reading, not compiled, not run on hardware
//
#include <hip/hip_runtime.h>

#define BATCH   8
#define PARTS   42
#define NPTS    16384
#define CHUNK   1024
#define NCHUNK  (NPTS / CHUNK)      // 16 chunks
#define NTHREAD 256
#define ITER    (CHUNK / (8 * 32))  // 4 iterations of 32 points per wave

typedef __attribute__((ext_vector_type(16))) __bf16        v16bf;
typedef __attribute__((ext_vector_type(8)))  float         v8f;
typedef __attribute__((ext_vector_type(8)))  unsigned int  v8u;

static __device__ __forceinline__ unsigned fbits(float f) {
    return __builtin_bit_cast(unsigned, f);
}

// pack two f32 into one dword {bf16(b) | bf16(a)} (a = low half), truncating
static __device__ __forceinline__ unsigned pkbf(float a, float b) {
    return __builtin_amdgcn_perm(fbits(b), fbits(a), 0x07060302u);
}

// pack then relu both bf16 halves with one packed max (avoids f32 canonicalize+max pairs)
static __device__ __forceinline__ unsigned pkrelu(float a, float b) {
    unsigned d = pkbf(a, b);
    unsigned r;
    asm("v_pk_max_num_bf16 %0, %1, 0" : "=v"(r) : "v"(d));
    return r;
}

static __device__ __forceinline__ v16bf asbf(v8u u) {
    return __builtin_bit_cast(v16bf, u);
}

// relu(D_lo,D_hi) packed straight into next layer's B fragment layout
static __device__ __forceinline__ v16bf packB(v8f lo, v8f hi) {
    v8u t;
    t[0] = pkrelu(lo[0], lo[1]); t[1] = pkrelu(lo[2], lo[3]);
    t[2] = pkrelu(lo[4], lo[5]); t[3] = pkrelu(lo[6], lo[7]);
    t[4] = pkrelu(hi[0], hi[1]); t[5] = pkrelu(hi[2], hi[3]);
    t[6] = pkrelu(hi[4], hi[5]); t[7] = pkrelu(hi[6], hi[7]);
    return asbf(t);
}

static __device__ __forceinline__ v8f wmma_bf16(v16bf a, v16bf b, v8f c) {
    return __builtin_amdgcn_wmma_f32_16x16x32_bf16(false, a, false, b, (short)0, c, false, false);
}

// fragments: 0,1 = layer0 (sigma_t0) ch-lo/hi ; 2,3 = layer0 (sigma_t1) ch-lo/hi
//            4,5 = layer1 lo/hi ; 6,7 = layer2 lo/hi ; 8 = layer3 (row0 = w3)
#define NFRAG 9

__global__ __launch_bounds__(NTHREAD) void sdf_mlp_kernel(
    const float* __restrict__ points,    // [B][P][N][3]
    const float* __restrict__ features,  // [B][P*8]
    const float* __restrict__ W0,        // [P][32][17]
    const float* __restrict__ W1,        // [P][32][32]
    const float* __restrict__ W2,        // [P][32][32]
    const float* __restrict__ W3,        // [P][1][32]
    float* __restrict__ out)             // [B][N][P]
{
    __shared__ __align__(32) unsigned short wfrag[NFRAG][32][16];

    const int chunk = blockIdx.x;
    const int p     = blockIdx.y;
    const int b     = blockIdx.z;
    const int tid   = threadIdx.x;

    const float* W0p = W0 + p * 32 * 17;
    const float* W1p = W1 + p * 32 * 32;
    const float* W2p = W2 + p * 32 * 32;
    const float* W3p = W3 + p * 32;

    // ---- one-time per block: build A-fragment-ready weights in LDS ----
    for (int e = tid; e < NFRAG * 32 * 16; e += NTHREAD) {
        int f    = e >> 9;           // fragment
        int lane = (e >> 4) & 31;
        int j    = e & 15;
        int half = lane >> 4;
        // A 16-bit 16x32 layout (contiguous k order): elem j -> K = 16*half+j
        int K = half * 16 + j;
        int pi = ((K & 8) << 1) | ((K & 16) >> 1) | (K & 7);  // swap bits 3<->4
        float v = 0.f;
        if (f < 2) {                 // layer0, tile0 ordering: dim = K (0..16)
            int outch = (f & 1) * 16 + (lane & 15);
            if (K < 17) v = W0p[outch * 17 + K];
        } else if (f < 4) {          // layer0, tile1 ordering: dim = K-16 (K>=16), dim16 at K==0
            int outch = (f & 1) * 16 + (lane & 15);
            int d = (K == 0) ? 16 : (K >= 16 ? K - 16 : -1);
            if (d >= 0) v = W0p[outch * 17 + d];
        } else if (f < 6) {          // layer1, pi-permuted K
            int outch = (f & 1) * 16 + (lane & 15);
            v = W1p[outch * 32 + pi];
        } else if (f < 8) {          // layer2, pi-permuted K
            int outch = (f & 1) * 16 + (lane & 15);
            v = W2p[outch * 32 + pi];
        } else {                     // layer3: only out-channel 0 is real
            int outch = lane & 15;
            if (outch == 0) v = W3p[pi];
        }
        unsigned bits = fbits(v);
        unsigned rne  = (bits + 0x7FFFu + ((bits >> 16) & 1u)) >> 16;  // RNE to bf16
        wfrag[f][lane][j] = (unsigned short)rne;
    }
    __syncthreads();

    const int lane = tid & 31;
    const int wave = tid >> 5;
    const int half = lane >> 4;

    v16bf a0lo_t0 = *(const v16bf*)wfrag[0][lane];
    v16bf a0hi_t0 = *(const v16bf*)wfrag[1][lane];
    v16bf a0lo_t1 = *(const v16bf*)wfrag[2][lane];
    v16bf a0hi_t1 = *(const v16bf*)wfrag[3][lane];
    v16bf a1lo    = *(const v16bf*)wfrag[4][lane];
    v16bf a1hi    = *(const v16bf*)wfrag[5][lane];
    v16bf a2lo    = *(const v16bf*)wfrag[6][lane];
    v16bf a2hi    = *(const v16bf*)wfrag[7][lane];
    v16bf a3      = *(const v16bf*)wfrag[8][lane];

    // features for (b,p): uniform
    const float* F = features + (b * PARTS + p) * 8;
    float    f0   = F[0];
    unsigned d5   = pkbf(F[1], F[2]);
    unsigned d6   = pkbf(F[3], F[4]);
    unsigned d7   = pkbf(F[5], F[6]);
    unsigned f7pk = pkbf(F[7], 0.f);

    const int    nbase = chunk * CHUNK;
    const float* PP    = points + (((size_t)b * PARTS + p) * NPTS + nbase) * 3;
    float*       OUT   = out + ((size_t)b * NPTS + nbase) * PARTS + p;

    const float C2PI = 0.15915494309189535f;  // 1/(2*pi) for v_sin/v_cos
    const v8f czero = {};

#pragma unroll 1
    for (int it = 0; it < ITER; ++it) {
        // this wave handles 32 consecutive points (2 WMMA tiles), one point per lane
        int n0 = (wave * ITER + it) * 32;
        const float* q = PP + (size_t)(n0 + lane) * 3;
        if (it + 1 < ITER) __builtin_prefetch(q + 32 * 3, 0, 0);
        float x = q[0], y = q[1], z = q[2];
        float sx = __builtin_amdgcn_sinf(x * C2PI);
        float sy = __builtin_amdgcn_sinf(y * C2PI);
        float sz = __builtin_amdgcn_sinf(z * C2PI);
        float cx = __builtin_amdgcn_cosf(x * C2PI);
        float cy = __builtin_amdgcn_cosf(y * C2PI);
        float cz = __builtin_amdgcn_cosf(z * C2PI);

        // P = this lane's 16 input dims [x,y,z,sx,sy,sz,cx,cy,cz,f0..f6]; U = [f7,0..0]
        unsigned P0 = pkbf(x,  y);
        unsigned P1 = pkbf(z,  sx);
        unsigned P2 = pkbf(sy, sz);
        unsigned P3 = pkbf(cx, cy);
        unsigned P4 = pkbf(cz, f0);

        // tile0 B frag: lanes<16 supply their own point dims (K0..15), lanes>=16 -> [f7,0..]
        // tile1 B frag: roles swapped (layer0 weights for tile1 are built K-shifted by 16)
        v8u b0d, b1d;
        b0d[0] = half ? f7pk : P0;   b1d[0] = half ? P0 : f7pk;
        b0d[1] = half ? 0u   : P1;   b1d[1] = half ? P1 : 0u;
        b0d[2] = half ? 0u   : P2;   b1d[2] = half ? P2 : 0u;
        b0d[3] = half ? 0u   : P3;   b1d[3] = half ? P3 : 0u;
        b0d[4] = half ? 0u   : P4;   b1d[4] = half ? P4 : 0u;
        b0d[5] = half ? 0u   : d5;   b1d[5] = half ? d5 : 0u;
        b0d[6] = half ? 0u   : d6;   b1d[6] = half ? d6 : 0u;
        b0d[7] = half ? 0u   : d7;   b1d[7] = half ? d7 : 0u;
        v16bf bm0 = asbf(b0d);
        v16bf bm1 = asbf(b1d);

        // ---- interleaved two-tile pipeline: each tile's transition VALU hides the
        // other tile's WMMA->VALU hazard slots ----
        v8f c0lo = wmma_bf16(a0lo_t0, bm0, czero);
        v8f c0hi = wmma_bf16(a0hi_t0, bm0, czero);
        v8f c1lo = wmma_bf16(a0lo_t1, bm1, czero);
        v8f c1hi = wmma_bf16(a0hi_t1, bm1, czero);

        v16bf b1_0 = packB(c0lo, c0hi);
        c0lo = wmma_bf16(a1lo, b1_0, czero);
        c0hi = wmma_bf16(a1hi, b1_0, czero);
        v16bf b1_1 = packB(c1lo, c1hi);
        c1lo = wmma_bf16(a1lo, b1_1, czero);
        c1hi = wmma_bf16(a1hi, b1_1, czero);

        v16bf b2_0 = packB(c0lo, c0hi);
        c0lo = wmma_bf16(a2lo, b2_0, czero);
        c0hi = wmma_bf16(a2hi, b2_0, czero);
        v16bf b2_1 = packB(c1lo, c1hi);
        c1lo = wmma_bf16(a2lo, b2_1, czero);
        c1hi = wmma_bf16(a2hi, b2_1, czero);

        // layer3 as WMMA: row 0 of A3 = w3, result lands in D[0] of lanes 0..15
        v16bf b3_0 = packB(c0lo, c0hi);
        c0lo = wmma_bf16(a3, b3_0, czero);
        v16bf b3_1 = packB(c1lo, c1hi);
        c1lo = wmma_bf16(a3, b3_1, czero);

        if (half == 0) {
            OUT[(size_t)(n0 + lane) * PARTS]      = c0lo[0];   // tile0 point
            OUT[(size_t)(n0 + 16 + lane) * PARTS] = c1lo[0];   // tile1 point
        }
    }
}

extern "C" void kernel_launch(void* const* d_in, const int* in_sizes, int n_in,
                              void* d_out, int out_size, void* d_ws, size_t ws_size,
                              hipStream_t stream) {
    (void)in_sizes; (void)n_in; (void)d_ws; (void)ws_size; (void)out_size;
    const float* points   = (const float*)d_in[0];
    const float* features = (const float*)d_in[1];
    const float* W0       = (const float*)d_in[2];
    const float* W1       = (const float*)d_in[3];
    const float* W2       = (const float*)d_in[4];
    const float* W3       = (const float*)d_in[5];
    float* out            = (float*)d_out;

    dim3 grid(NCHUNK, PARTS, BATCH);   // 16 x 42 x 8 = 5376 blocks
    sdf_mlp_kernel<<<grid, NTHREAD, 0, stream>>>(points, features, W0, W1, W2, W3, out);
}
